// RBFClassifier_23922967839460
// MI455X (gfx1250) — matrix operand order repaired and checked by reading, not compiled, not run on hardware
//
#include <hip/hip_runtime.h>
#include <stdint.h>

// ---------------------------------------------------------------------------
// RBF classifier fused for MI455X (gfx1250, wave32, WMMA + async-to-LDS).
//   d2  = ||x||^2 + ||c||^2 - 2 x.c           (cross term: v_wmma bf16)
//   phi = exp(-max(d2,0) * exp(-2 log_sigma))  (sqrt cancels vs square)
//   out = phi @ W^T + b                        (folded, per-row LDS accum)
// B=16384, D=784 (bf16-padded to 800), C=2048, OUT=10.
// Double-buffered LDS tiles filled by global_load_async_to_lds_b128 issued one
// k-step ahead (ASYNCcnt), so DMA overlaps the WMMA stream.
// Workspace: x_bf16 (25 MB) + c_bf16 (3.1 MB) + norms (~80 KB) ~= 29.6 MB.
// ---------------------------------------------------------------------------

#define BD   784
#define DP   800
#define CC   2048
#define NOUT 10
#define NB   16384
#define BM   64          // x rows per workgroup
#define BN   128         // centres per column chunk
#define KC   32          // WMMA bf16 K step
#define NKC  25          // DP / KC
#define NTN  (CC / BN)   // 16
#define LSTR 40          // LDS tile row stride in halves (80 B, 16 B aligned)
#define PSTR (BN + 5)    // phi row stride (133: gcd(133,64)=1, conflict-free)

// dynamic LDS partition sizes (elements)
#define ASZ  (BM * LSTR)          // halves per A buffer
#define BSZ  (BN * LSTR)          // halves per B buffer
#define SMEM_BYTES ((2 * ASZ + 2 * BSZ) * 2 + \
                    (BM * PSTR + NOUT * BN + BM * NOUT + BM) * 4)   // 72,704 B

typedef __attribute__((ext_vector_type(16))) __bf16 v16bf;
typedef __attribute__((ext_vector_type(8)))  float  v8f;
typedef __attribute__((ext_vector_type(4))) unsigned int u32x4;

union Frag { u32x4 q[2]; v16bf v; };

#define WMMA_BF16(A, B, C) \
    __builtin_amdgcn_wmma_f32_16x16x32_bf16(false, (A), false, (B), (short)0, (C), false, false)

// ---------------------------------------------------------------------------
// Prep 1: x -> bf16 (padded, stride DP) + ||x||^2. One block per row.
// ---------------------------------------------------------------------------
__global__ __launch_bounds__(256)
void rbf_prep_x(const float* __restrict__ x, __bf16* __restrict__ xbf,
                float* __restrict__ x2) {
    __shared__ float red[256];
    const int row = blockIdx.x;
    const float* src = x + (size_t)row * BD;
    __bf16* dst = xbf + (size_t)row * DP;
    float s = 0.f;
    for (int d = threadIdx.x; d < DP; d += 256) {
        const float v = (d < BD) ? src[d] : 0.f;
        dst[d] = (__bf16)v;
        s += v * v;
    }
    red[threadIdx.x] = s;
    __syncthreads();
    for (int w = 128; w > 0; w >>= 1) {
        if ((int)threadIdx.x < w) red[threadIdx.x] += red[threadIdx.x + w];
        __syncthreads();
    }
    if (threadIdx.x == 0) x2[row] = red[0];
}

// ---------------------------------------------------------------------------
// Prep 2: centres -> bf16 + ||c||^2 + exp(-2 log_sigma). One block per centre.
// ---------------------------------------------------------------------------
__global__ __launch_bounds__(256)
void rbf_prep_c(const float* __restrict__ centres, const float* __restrict__ log_sigmas,
                __bf16* __restrict__ cbf, float* __restrict__ c2,
                float* __restrict__ is2) {
    __shared__ float red[256];
    const int row = blockIdx.x;
    const float* src = centres + (size_t)row * BD;
    __bf16* dst = cbf + (size_t)row * DP;
    float s = 0.f;
    for (int d = threadIdx.x; d < DP; d += 256) {
        const float v = (d < BD) ? src[d] : 0.f;
        dst[d] = (__bf16)v;
        s += v * v;
    }
    red[threadIdx.x] = s;
    __syncthreads();
    for (int w = 128; w > 0; w >>= 1) {
        if ((int)threadIdx.x < w) red[threadIdx.x] += red[threadIdx.x + w];
        __syncthreads();
    }
    if (threadIdx.x == 0) {
        c2[row]  = red[0];
        is2[row] = __expf(-2.0f * log_sigmas[row]);
    }
}

// ---------------------------------------------------------------------------
// Main fused kernel. Grid NB/BM = 256 WGs x 256 threads (8 waves).
// Wave tile: 2x2 WMMA tiles -> 4 wmma / 8 ds_load_b128 per k-step.
// ---------------------------------------------------------------------------
__global__ __launch_bounds__(256)
void rbf_main(const __bf16* __restrict__ xbf, const __bf16* __restrict__ cbf,
              const float* __restrict__ x2g, const float* __restrict__ c2g,
              const float* __restrict__ is2g, const float* __restrict__ Wg,
              const float* __restrict__ bg, float* __restrict__ outg) {
    extern __shared__ __align__(16) unsigned char smem[];
    unsigned short* As   = (unsigned short*)smem;        // 2 x ASZ halves
    unsigned short* Bs   = As + 2 * ASZ;                 // 2 x BSZ halves
    float* phiB   = (float*)(Bs + 2 * BSZ);              // BM x PSTR
    float* Ws     = phiB + BM * PSTR;                    // NOUT x BN
    float* outAcc = Ws + NOUT * BN;                      // BM x NOUT
    float* x2s    = outAcc + BM * NOUT;                  // BM

    const int tid  = threadIdx.x;
    const int lane = tid & 31;
    const int wave = tid >> 5;
    const int grp  = lane >> 4;       // lane group: 0 = lanes 0-15, 1 = 16-31
    const int l16  = lane & 15;
    const int rowBase = blockIdx.x * BM;

    for (int i = tid; i < BM * NOUT; i += 256) outAcc[i] = 0.f;
    if (tid < BM) x2s[tid] = x2g[rowBase + tid];

    // ---- async staging addresses (16 B chunk per lane per issue) ----
    const int ar = tid >> 2, aq = tid & 3;        // A: 64 rows x 4 chunks
    const int br = tid >> 1, bq = (tid & 1) * 2;  // B: 128 rows x 4 chunks (2/thread)
    const unsigned ldsA0 = (unsigned)(uintptr_t)As + (unsigned)(ar * LSTR * 2 + aq * 16);
    const unsigned ldsB0 = (unsigned)(uintptr_t)Bs + (unsigned)(br * LSTR * 2 + bq * 16);
    const unsigned long long gA0 =
        (unsigned long long)(uintptr_t)(xbf + (size_t)(rowBase + ar) * DP + aq * 8);

    const int tmBase = (wave >> 2) * 2;   // {0,2}:     A tile rows
    const int tnBase = (wave & 3) * 2;    // {0,2,4,6}: B tile cols

    __syncthreads();

    for (int tn = 0; tn < NTN; ++tn) {
        v8f acc00 = {}, acc01 = {}, acc10 = {}, acc11 = {};
        const unsigned long long gB0 =
            (unsigned long long)(uintptr_t)(cbf + (size_t)(tn * BN + br) * DP + bq * 8);

        // ---- pipeline prologue: fill buffer 0 with k-chunk 0 ----
        // (previous tn's trailing barrier guarantees nobody still reads buf0)
        asm volatile("global_load_async_to_lds_b128 %0, %1, off"
                     :: "v"(ldsA0), "v"(gA0) : "memory");
        asm volatile("global_load_async_to_lds_b128 %0, %1, off"
                     :: "v"(ldsB0), "v"(gB0) : "memory");
        asm volatile("global_load_async_to_lds_b128 %0, %1, off offset:16"
                     :: "v"(ldsB0), "v"(gB0) : "memory");

#pragma unroll 1
        for (int kc = 0; kc < NKC; ++kc) {
            const int cur = kc & 1;
            // drain own async issues for buffer `cur`, then publish to all waves
            asm volatile("s_wait_asynccnt 0x0" ::: "memory");
            __syncthreads();

            // issue next k-chunk into the other buffer (overlaps with WMMA below;
            // barrier above proves all waves finished reading it last step)
            if (kc + 1 < NKC) {
                const int nxt = cur ^ 1;
                const unsigned long long ga = gA0 + (unsigned long long)((kc + 1) * KC * 2);
                const unsigned long long gb = gB0 + (unsigned long long)((kc + 1) * KC * 2);
                const unsigned la = ldsA0 + (unsigned)(nxt * ASZ * 2);
                const unsigned lb = ldsB0 + (unsigned)(nxt * BSZ * 2);
                asm volatile("global_load_async_to_lds_b128 %0, %1, off"
                             :: "v"(la), "v"(ga) : "memory");
                asm volatile("global_load_async_to_lds_b128 %0, %1, off"
                             :: "v"(lb), "v"(gb) : "memory");
                asm volatile("global_load_async_to_lds_b128 %0, %1, off offset:16"
                             :: "v"(lb), "v"(gb) : "memory");
            }

            // ---- fragments per CDNA5 WMMA VGPR layouts ----
            const unsigned short* Asb = As + cur * ASZ;
            const unsigned short* Bsb = Bs + cur * BSZ;
            Frag fa0, fa1, fb0, fb1;
            {   // A 16x32: halves 0-7 <- K=8*grp.., halves 8-15 <- K=16+8*grp..
                const unsigned short* a0 = Asb + (tmBase * 16 + l16) * LSTR;
                const unsigned short* a1 = Asb + ((tmBase + 1) * 16 + l16) * LSTR;
                fa0.q[0] = *(const u32x4*)(a0 + 8 * grp);
                fa0.q[1] = *(const u32x4*)(a0 + 16 + 8 * grp);
                fa1.q[0] = *(const u32x4*)(a1 + 8 * grp);
                fa1.q[1] = *(const u32x4*)(a1 + 16 + 8 * grp);
                // B 32x16: halves 0-7 <- K=8*grp.., halves 8-15 <- K=16+8*grp.., N = l16
                const unsigned short* b0 = Bsb + (tnBase * 16 + l16) * LSTR;
                const unsigned short* b1 = Bsb + ((tnBase + 1) * 16 + l16) * LSTR;
                fb0.q[0] = *(const u32x4*)(b0 + 8 * grp);
                fb0.q[1] = *(const u32x4*)(b0 + 16 + 8 * grp);
                fb1.q[0] = *(const u32x4*)(b1 + 8 * grp);
                fb1.q[1] = *(const u32x4*)(b1 + 16 + 8 * grp);
            }
            acc00 = WMMA_BF16(fa0.v, fb0.v, acc00);
            acc01 = WMMA_BF16(fa0.v, fb1.v, acc01);
            acc10 = WMMA_BF16(fa1.v, fb0.v, acc10);
            acc11 = WMMA_BF16(fa1.v, fb1.v, acc11);
        }

        // ---- epilogue: S -> phi (C/D layout: M = j + 8*grp, N = l16) ----
#pragma unroll
        for (int j = 0; j < 8; ++j) {
            const int m0  = tmBase * 16 + 8 * grp + j;
            const int m1  = m0 + 16;
            const int nl0 = tnBase * 16 + l16;
            const int nl1 = nl0 + 16;
            const int ng0 = tn * BN + nl0;
            const int ng1 = tn * BN + nl1;
            const float c20 = c2g[ng0], c21 = c2g[ng1];
            const float i20 = is2g[ng0], i21 = is2g[ng1];
            const float xm0 = x2s[m0], xm1 = x2s[m1];
            phiB[m0 * PSTR + nl0] = __expf(-fmaxf(xm0 + c20 - 2.f * acc00[j], 0.f) * i20);
            phiB[m0 * PSTR + nl1] = __expf(-fmaxf(xm0 + c21 - 2.f * acc01[j], 0.f) * i21);
            phiB[m1 * PSTR + nl0] = __expf(-fmaxf(xm1 + c20 - 2.f * acc10[j], 0.f) * i20);
            phiB[m1 * PSTR + nl1] = __expf(-fmaxf(xm1 + c21 - 2.f * acc11[j], 0.f) * i21);
        }
        // stage this chunk's W slice (10 x 128) into LDS once
        for (int p = tid; p < NOUT * BN; p += 256) {
            const int o = p >> 7, n = p & (BN - 1);
            Ws[o * BN + n] = Wg[(size_t)o * CC + tn * BN + n];
        }
        __syncthreads();

        // ---- tiny second GEMM: outAcc[m,o] += phi[m,:] . W[o, chunk] ----
        for (int p = tid; p < BM * NOUT; p += 256) {
            const int m = p & 63;
            const int o = p >> 6;
            const float* wrow = Ws + o * BN;
            const float* prow = phiB + m * PSTR;
            float s = 0.f;
#pragma unroll 8
            for (int n = 0; n < BN; ++n) s += prow[n] * wrow[n];
            outAcc[p] += s;
        }
        __syncthreads();   // outAcc/phiB consumed before next tn reuses LDS
    }

    for (int p = tid; p < BM * NOUT; p += 256) {
        const int m = p & 63;
        const int o = p >> 6;
        outg[(size_t)(rowBase + m) * NOUT + o] = outAcc[p] + bg[o];
    }
}

// ---------------------------------------------------------------------------
extern "C" void kernel_launch(void* const* d_in, const int* in_sizes, int n_in,
                              void* d_out, int out_size, void* d_ws, size_t ws_size,
                              hipStream_t stream) {
    const float* x          = (const float*)d_in[0];
    const float* centres    = (const float*)d_in[1];
    const float* log_sigmas = (const float*)d_in[2];
    const float* W          = (const float*)d_in[3];
    const float* b          = (const float*)d_in[4];
    float* out = (float*)d_out;

    // workspace layout (~29.6 MB)
    __bf16* xbf = (__bf16*)d_ws;                       // NB*DP halves
    __bf16* cbf = xbf + (size_t)NB * DP;               // CC*DP halves
    float*  x2  = (float*)(cbf + (size_t)CC * DP);     // NB floats
    float*  c2  = x2 + NB;                             // CC floats
    float*  is2 = c2 + CC;                             // CC floats

    rbf_prep_x<<<NB, 256, 0, stream>>>(x, xbf, x2);
    rbf_prep_c<<<CC, 256, 0, stream>>>(centres, log_sigmas, cbf, c2, is2);
    rbf_main<<<NB / BM, 256, SMEM_BYTES, stream>>>(xbf, cbf, x2, c2, is2, W, b, out);
}
